// GroupedQueryAttention_58497454571567
// MI455X (gfx1250) — matrix operand order repaired and checked by reading, not compiled, not run on hardware
//
#include <hip/hip_runtime.h>
#include <cstdint>

typedef __attribute__((ext_vector_type(16))) __bf16 v16bf;
typedef __attribute__((ext_vector_type(8)))  float  v8f;

#define B_    2
#define T_    2048
#define C_    2048
#define H_    16
#define KV_   4
#define D_    128
#define QKVN_ 3072
#define M_    (B_*T_)

union BF16x16 { v16bf v; uint32_t u[8]; };

// A-fragment (16x32 bf16) K-pair base offsets within a 32-wide K slice
__device__ __forceinline__ int a_kofs(int p) { return (p < 4) ? 2*p : 2*p + 8; }

// Async global->LDS copy of 16B per lane (ASYNCcnt-tracked, bypasses VGPRs)
__device__ __forceinline__ void async_ld_b128(uint32_t lds_addr, const void* gaddr) {
  asm volatile("global_load_async_to_lds_b128 %0, %1, off"
               :: "v"(lds_addr), "v"(gaddr) : "memory");
}

// ---------------------------------------------------------------- casts
__global__ void __launch_bounds__(256)
cast_f32_bf16(const float* __restrict__ s, __bf16* __restrict__ d, int n) {
  int i = blockIdx.x * 256 + threadIdx.x;
  if (i < n) d[i] = (__bf16)s[i];
}

// ------------------------------------------------ NT GEMM: C = A * W^T
// A: [M,K] bf16 row-major, W: [N,K] bf16 row-major, C: [M,N] f32.
// Block = 128 threads (4 waves); wave w computes rows [m0, m0+16),
// block covers cols [n0, n0+128). W tiles double-buffered in LDS via
// global_load_async_to_lds_b128.
__global__ void __launch_bounds__(128)
gemm_nt_bf16(const __bf16* __restrict__ A, const __bf16* __restrict__ W,
             float* __restrict__ C, int M, int N, int K) {
  __shared__ __attribute__((aligned(16))) __bf16 bs[2][128][40]; // [buf][n][k] padded
  const int lane = threadIdx.x & 31;
  const int wave = threadIdx.x >> 5;
  const int hf   = lane >> 4;
  const int l15  = lane & 15;
  const int n0   = blockIdx.x * 128;
  const int m0   = (blockIdx.y * 4 + wave) * 16;

  v8f acc[8] = {};
  const __bf16* arow = A + (size_t)(m0 + l15) * K;
  const __bf16* wrow = W + (size_t)(n0 + threadIdx.x) * K;  // this thread's W row

  // prologue: stage k-slice 0 into buffer 0 (4 async b128 ops/wave)
  {
    uint32_t ldst = (uint32_t)(uintptr_t)&bs[0][threadIdx.x][0];
    const char* g = (const char*)wrow;
#pragma unroll
    for (int c = 0; c < 4; ++c) async_ld_b128(ldst + 16 * c, g + 16 * c);
  }

  const int nk = K / 32;
  for (int ki = 0; ki < nk; ++ki) {
    const int cur = ki & 1;
    if (ki + 1 < nk) {            // prefetch next slice into other buffer
      uint32_t ldst = (uint32_t)(uintptr_t)&bs[cur ^ 1][threadIdx.x][0];
      const char* g = (const char*)(wrow + (ki + 1) * 32);
#pragma unroll
      for (int c = 0; c < 4; ++c) async_ld_b128(ldst + 16 * c, g + 16 * c);
      asm volatile("s_wait_asynccnt 0x4" ::: "memory"); // cur's 4 landed
    } else {
      asm volatile("s_wait_asynccnt 0x0" ::: "memory");
    }
    __syncthreads();

    BF16x16 af;
#pragma unroll
    for (int p = 0; p < 8; ++p)
      af.u[p] = *(const uint32_t*)(arow + ki * 32 + 2 * p + 16 * hf);

    BF16x16 bw[8];                // preload all B fragments, then 8 WMMAs
#pragma unroll
    for (int j = 0; j < 8; ++j) {
      const __bf16* brow = &bs[cur][16 * j + l15][0];
#pragma unroll
      for (int p = 0; p < 8; ++p)
        bw[j].u[p] = *(const uint32_t*)(brow + 2 * p + 16 * hf);
    }
#pragma unroll
    for (int j = 0; j < 8; ++j)
      acc[j] = __builtin_amdgcn_wmma_f32_16x16x32_bf16(
          false, af.v, false, bw[j].v, (short)0, acc[j], false, false);
    __syncthreads();              // all waves done reading cur before reuse
  }
#pragma unroll
  for (int j = 0; j < 8; ++j) {
    int n = n0 + 16 * j + l15;
#pragma unroll
    for (int r = 0; r < 8; ++r) {
      int m = m0 + r + 8 * hf;
      C[(size_t)m * N + n] = acc[j][r];
    }
  }
}

// ------------------------------- RoPE (then) RMSNorm epilogue, split q/k/v
// grid (B*T, 24 heads), block 128 (one thread per dim)
__global__ void __launch_bounds__(128)
rope_norm_kernel(const float* __restrict__ qkv,
                 const float* __restrict__ fc, const float* __restrict__ fs,
                 const float* __restrict__ qw, const float* __restrict__ kw,
                 __bf16* __restrict__ qo, __bf16* __restrict__ ko,
                 __bf16* __restrict__ vo) {
  const int bt = blockIdx.x;
  const int head = blockIdx.y;
  const int b = bt / T_, t = bt % T_;
  const int d = threadIdx.x;
  float x = qkv[(size_t)bt * QKVN_ + head * D_ + d];
  float out;
  if (head < H_ + KV_) {               // q or k: rope + rmsnorm
    int i = d >> 1;
    float c = fc[t * (D_ / 2) + i];
    float s = fs[t * (D_ / 2) + i];
    float partner = __shfl_xor(x, 1, 32);
    float val = (d & 1) ? (partner * s + x * c) : (x * c - partner * s);
    float ss = val * val;
#pragma unroll
    for (int m = 1; m < 32; m <<= 1) ss += __shfl_xor(ss, m, 32);
    __shared__ float part[4];
    if ((threadIdx.x & 31) == 0) part[threadIdx.x >> 5] = ss;
    __syncthreads();
    float tot = part[0] + part[1] + part[2] + part[3];
    float rms = sqrtf(tot * (1.0f / D_) + 1e-6f);
    const float* w = (head < H_) ? qw : kw;
    out = w[d] * (val / rms);
  } else {
    out = x;                           // v: passthrough
  }
  if (head < H_) {
    qo[(((size_t)(b * H_ + head)) * T_ + t) * D_ + d] = (__bf16)out;
  } else if (head < H_ + KV_) {
    ko[(((size_t)(b * KV_ + (head - H_))) * T_ + t) * D_ + d] = (__bf16)out;
  } else {
    vo[(((size_t)(b * KV_ + (head - H_ - KV_))) * T_ + t) * D_ + d] = (__bf16)out;
  }
}

// ------------------------------------- flash attention, 1 wave / q-tile
// grid (T/16, B*H), block 32
__global__ void __launch_bounds__(32)
attn_kernel(const __bf16* __restrict__ q, const __bf16* __restrict__ k,
            const __bf16* __restrict__ v, __bf16* __restrict__ y) {
  const int qt = blockIdx.x;
  const int bh = blockIdx.y;
  const int b = bh / H_, h = bh % H_, kvh = h / (H_ / KV_);
  const int lane = threadIdx.x & 31;
  const int hf = lane >> 4, l15 = lane & 15;
  const int q0 = qt * 16;

  const __bf16* qp = q + (size_t)(b * H_ + h) * T_ * D_;
  const __bf16* kp = k + (size_t)(b * KV_ + kvh) * T_ * D_;
  const __bf16* vp = v + (size_t)(b * KV_ + kvh) * T_ * D_;

  __shared__ __attribute__((aligned(16))) __bf16 pbuf[16][40]; // P: 16 rows x 32 keys

  BF16x16 aq[4];                       // Q tile as 4 A-fragments (16x128)
#pragma unroll
  for (int j = 0; j < 4; ++j)
#pragma unroll
    for (int p = 0; p < 8; ++p)
      aq[j].u[p] = *(const uint32_t*)(qp + (size_t)(q0 + l15) * D_ +
                                      32 * j + 2 * p + 16 * hf);

  v8f acc[8] = {};
  float rmax[8], rsum[8];
#pragma unroll
  for (int r = 0; r < 8; ++r) { rmax[r] = -1e30f; rsum[r] = 0.f; }

  const int nch = (qt + 2) >> 1;       // ceil((qt+1)/2) 32-key chunks
  for (int ch = 0; ch < nch; ++ch) {
    const int kb = ch * 32;
    v8f s0 = {}, s1 = {};
#pragma unroll
    for (int j = 0; j < 4; ++j) {      // S tile 0 (keys kb..kb+15)
      BF16x16 bk;
#pragma unroll
      for (int p = 0; p < 8; ++p)
        bk.u[p] = *(const uint32_t*)(kp + (size_t)(kb + l15) * D_ +
                                     32 * j + 2 * p + 16 * hf);
      s0 = __builtin_amdgcn_wmma_f32_16x16x32_bf16(
          false, aq[j].v, false, bk.v, (short)0, s0, false, false);
    }
#pragma unroll
    for (int j = 0; j < 4; ++j) {      // S tile 1 (keys kb+16..kb+31)
      BF16x16 bk;
#pragma unroll
      for (int p = 0; p < 8; ++p)
        bk.u[p] = *(const uint32_t*)(kp + (size_t)(kb + 16 + l15) * D_ +
                                     32 * j + 2 * p + 16 * hf);
      s1 = __builtin_amdgcn_wmma_f32_16x16x32_bf16(
          false, aq[j].v, false, bk.v, (short)0, s1, false, false);
    }
    const float scale = 0.08838834764831845f; // 1/sqrt(128)
    float p0[8], p1[8];
#pragma unroll
    for (int r = 0; r < 8; ++r) {
      int qrow = q0 + r + 8 * hf;
      float a0 = s0[r] * scale; if (kb + l15      > qrow) a0 = -1e30f;
      float a1 = s1[r] * scale; if (kb + 16 + l15 > qrow) a1 = -1e30f;
      float cm = fmaxf(a0, a1);
#pragma unroll
      for (int m = 1; m < 16; m <<= 1) cm = fmaxf(cm, __shfl_xor(cm, m, 32));
      float nm = fmaxf(rmax[r], cm);
      float alpha = __expf(rmax[r] - nm);
      float e0 = __expf(a0 - nm);
      float e1 = __expf(a1 - nm);
      float ls = e0 + e1;
#pragma unroll
      for (int m = 1; m < 16; m <<= 1) ls += __shfl_xor(ls, m, 32);
      rsum[r] = rsum[r] * alpha + ls;
      rmax[r] = nm;
      p0[r] = e0; p1[r] = e1;
#pragma unroll
      for (int j = 0; j < 8; ++j) acc[j][r] *= alpha;
    }
    __syncthreads();                   // protect pbuf WAR
#pragma unroll
    for (int r = 0; r < 8; ++r) {      // D-frag -> LDS (row-major P)
      pbuf[r + 8 * hf][l15]      = (__bf16)p0[r];
      pbuf[r + 8 * hf][16 + l15] = (__bf16)p1[r];
    }
    __syncthreads();
    BF16x16 ap;                        // reload P as A-fragment (16x32)
#pragma unroll
    for (int p = 0; p < 8; ++p)
      ap.u[p] = *(const uint32_t*)(&pbuf[l15][2 * p + 16 * hf]);
#pragma unroll
    for (int j = 0; j < 8; ++j) {      // acc += P * V (contract over keys)
      BF16x16 bv;
#pragma unroll
      for (int p = 0; p < 8; ++p) {
        int kk = kb + 2 * p + 16 * hf;
        int n  = 16 * j + l15;
        uint32_t lo = *(const uint16_t*)(vp + (size_t)kk * D_ + n);
        uint32_t hi = *(const uint16_t*)(vp + (size_t)(kk + 1) * D_ + n);
        bv.u[p] = lo | (hi << 16);
      }
      acc[j] = __builtin_amdgcn_wmma_f32_16x16x32_bf16(
          false, ap.v, false, bv.v, (short)0, acc[j], false, false);
    }
  }
#pragma unroll
  for (int j = 0; j < 8; ++j) {        // normalize + write y[b][t][h*128+d]
#pragma unroll
    for (int r = 0; r < 8; ++r) {
      int m = r + 8 * hf;
      float o = acc[j][r] / rsum[r];
      y[((size_t)b * T_ + q0 + m) * C_ + h * D_ + 16 * j + l15] = (__bf16)o;
    }
  }
}

// ---------------------------------------------------------------- launch
extern "C" void kernel_launch(void* const* d_in, const int* in_sizes, int n_in,
                              void* d_out, int out_size, void* d_ws, size_t ws_size,
                              hipStream_t stream) {
  (void)in_sizes; (void)n_in; (void)out_size; (void)ws_size;
  const float* x     = (const float*)d_in[0];
  const float* wqkv  = (const float*)d_in[1];
  const float* wproj = (const float*)d_in[2];
  const float* qw    = (const float*)d_in[3];
  const float* kw    = (const float*)d_in[4];
  const float* fc    = (const float*)d_in[5];
  const float* fs    = (const float*)d_in[6];
  float* out = (float*)d_out;

  char* w = (char*)d_ws;
  __bf16* xb     = (__bf16*)w; w += (size_t)M_ * C_ * 2;
  __bf16* wqkvb  = (__bf16*)w; w += (size_t)QKVN_ * C_ * 2;
  __bf16* wprojb = (__bf16*)w; w += (size_t)C_ * C_ * 2;
  float*  qkvf   = (float*)w;  w += (size_t)M_ * QKVN_ * 4;
  __bf16* qb     = (__bf16*)w; w += (size_t)B_ * H_ * T_ * D_ * 2;
  __bf16* kb     = (__bf16*)w; w += (size_t)B_ * KV_ * T_ * D_ * 2;
  __bf16* vb     = (__bf16*)w; w += (size_t)B_ * KV_ * T_ * D_ * 2;
  __bf16* yb     = (__bf16*)w; w += (size_t)M_ * C_ * 2;

  int nx = M_ * C_, nq = QKVN_ * C_, np = C_ * C_;
  cast_f32_bf16<<<(nx + 255) / 256, 256, 0, stream>>>(x, xb, nx);
  cast_f32_bf16<<<(nq + 255) / 256, 256, 0, stream>>>(wqkv, wqkvb, nq);
  cast_f32_bf16<<<(np + 255) / 256, 256, 0, stream>>>(wproj, wprojb, np);

  gemm_nt_bf16<<<dim3(QKVN_ / 128, M_ / 64), 128, 0, stream>>>(
      xb, wqkvb, qkvf, M_, QKVN_, C_);
  rope_norm_kernel<<<dim3(M_, H_ + 2 * KV_), 128, 0, stream>>>(
      qkvf, fc, fs, qw, kw, qb, kb, vb);
  attn_kernel<<<dim3(T_ / 16, B_ * H_), 32, 0, stream>>>(qb, kb, vb, yb);
  gemm_nt_bf16<<<dim3(C_ / 128, M_ / 64), 128, 0, stream>>>(
      yb, wprojb, out, M_, C_, C_);
}
